// GeneralNetworkedAE_79053168050863
// MI455X (gfx1250) — matrix operand order repaired and checked by reading, not compiled, not run on hardware
//
#include <hip/hip_runtime.h>
#include <hip/hip_bf16.h>

typedef __attribute__((ext_vector_type(16))) __bf16        v16bf;
typedef __attribute__((ext_vector_type(8)))  float         v8f;
typedef __attribute__((ext_vector_type(4)))  unsigned int  u32x4;

union Frag {
    v16bf v;
    u32x4 q[2];
};

// float -> bf16 bits, round-half-up (scalar, prep kernels)
__device__ __forceinline__ unsigned int f2bf(float f) {
    return (__float_as_uint(f) + 0x8000u) >> 16;
}

// two floats -> packed bf16x2: two rounding adds + one v_perm_b32
__device__ __forceinline__ unsigned int pack_bf16(float lo, float hi) {
    unsigned int a = __float_as_uint(lo) + 0x8000u;
    unsigned int b = __float_as_uint(hi) + 0x8000u;
    // result bytes {b3,b2,a3,a2} = {bf16(hi), bf16(lo)}
    return __builtin_amdgcn_perm(b, a, 0x07060302u);
}

__device__ __forceinline__ v8f wmma_bf16(v16bf a, v16bf b, v8f c) {
    // (neg_a, A, neg_b, B, c_mod, C, reuse_a, reuse_b)
    return __builtin_amdgcn_wmma_f32_16x16x32_bf16(false, a, false, b, (short)0, c, false, false);
}

// ---------------------------------------------------------------------------
// Problem constants (fixed by the reference)
#define B_TOT  16384
#define NX     896
#define NU     128
#define NA     32
#define DIN    128
#define HID    256
#define DOUT   28
#define DOUTP  32      // padded to 2 N-tiles

#define BTILE  128     // batch rows per block
#define GPE    136     // G LDS pitch in ushorts (272B -> stride 4 banks mod 64)
#define HPE    264     // H LDS pitch in ushorts (528B -> stride 4 banks mod 64)

// ---------------------------------------------------------------------------
// Prep: W1 (A,DIN,H) f32 -> W1t (A,H,DIN) bf16      total 32*256*128 = 1,048,576
__global__ void prep_w1(const float* __restrict__ W1, unsigned short* __restrict__ W1t) {
    int e = blockIdx.x * 256 + threadIdx.x;          // [a][n][k], k fastest
    int a = e >> 15;
    int r = e & 32767;
    int n = r >> 7;
    int k = r & 127;
    W1t[e] = (unsigned short)f2bf(W1[((size_t)a * DIN + k) * HID + n]);
}

// Prep: W2 (A,H,DOUT) f32 -> W2t (A,DOUTP,Hperm) bf16, zero-padded.
// K axis permuted to match packed H storage: within each 32-group,
// storage slot w holds original k = g*32 + (w>>1) + (w&1)*16.
__global__ void prep_w2(const float* __restrict__ W2, unsigned short* __restrict__ W2t) {
    int e = blockIdx.x * 256 + threadIdx.x;          // [a][o][s], s fastest
    int a = e >> 13;
    int r = e & 8191;
    int o = r >> 8;
    int s = r & 255;
    int g = s >> 5, w = s & 31;
    int k = g * 32 + (w >> 1) + ((w & 1) << 4);      // inverse of pack permutation
    W2t[e] = (o < DOUT) ? (unsigned short)f2bf(W2[((size_t)a * HID + k) * DOUT + o])
                        : (unsigned short)0;
}

// ---------------------------------------------------------------------------
// Fused per-head MLP: gather -> WMMA GEMM1 (+bias+relu) -> WMMA GEMM2 (+bias) -> out
__global__ __launch_bounds__(256) void mlp_fused(
    const float* __restrict__ x, const float* __restrict__ u,
    const unsigned short* __restrict__ W1t, const float* __restrict__ b1,
    const unsigned short* __restrict__ W2t, const float* __restrict__ b2,
    const int* __restrict__ in_idx, float* __restrict__ out)
{
    extern __shared__ char smem[];
    int*            sIdx = (int*)smem;                                    // 512 B
    unsigned short* G    = (unsigned short*)(smem + 512);                 // 128*272 B
    unsigned short* H    = (unsigned short*)(smem + 512 + BTILE * GPE * 2);
    unsigned int*   G32  = (unsigned int*)G;
    unsigned int*   H32  = (unsigned int*)H;

    const int a    = blockIdx.y;
    const int b0   = blockIdx.x * BTILE;
    const int tid  = threadIdx.x;
    const int wave = tid >> 5;
    const int lane = tid & 31;
    const int lh   = lane & 15;     // N (or M) within tile
    const int hi   = lane >> 4;     // half-wave select
    const int mg   = wave >> 2;     // M-group: owns m-tiles mg*4 .. mg*4+3
    const int ng   = wave & 3;      // N-group: owns n-tiles ng*4 .. ng*4+3

    const unsigned short* w1 = W1t + (size_t)a * HID * DIN;
    const unsigned short* w2 = W2t + (size_t)a * DOUTP * HID;
    __builtin_prefetch(w2, 0, 3);   // global_prefetch_b8: W2 is L2-resident

    if (tid < DIN) sIdx[tid] = in_idx[a * DIN + tid];

    // --- preload layer-1 B fragments into registers (per-wave: 4 N-tiles x 4 K-steps)
    Frag Bf[4][4];
#pragma unroll
    for (int nt = 0; nt < 4; ++nt) {
        int n = (ng * 4 + nt) * 16 + lh;
#pragma unroll
        for (int kf = 0; kf < 4; ++kf) {
            const unsigned short* p = w1 + n * DIN + kf * 32 + hi * 16;   // contiguous 32B
            Bf[nt][kf].q[0] = *(const u32x4*)p;
            Bf[nt][kf].q[1] = *(const u32x4*)(p + 8);
        }
    }
    __syncthreads();

    // --- gather: lane owns column pair (2*i2, 2*i2+1) for rows r0, r0+4, ...
    //     selects/addresses hoisted out of the loop (i2 invariant: 256 % 64 == 0)
    {
        const int i2 = tid & 63;
        const int r0 = tid >> 6;
        int i0 = sIdx[2 * i2], i1 = sIdx[2 * i2 + 1];
        const float* p0 = (i0 < NX) ? x + (size_t)(b0 + r0) * NX + i0
                                    : u + (size_t)(b0 + r0) * NU + (i0 - NX);
        const float* p1 = (i1 < NX) ? x + (size_t)(b0 + r0) * NX + i1
                                    : u + (size_t)(b0 + r0) * NU + (i1 - NX);
        const int st0 = (i0 < NX) ? 4 * NX : 4 * NU;   // elements per +4 rows
        const int st1 = (i1 < NX) ? 4 * NX : 4 * NU;
        unsigned int* gq = G32 + r0 * (GPE / 2) + i2;
#pragma unroll 8
        for (int it = 0; it < BTILE / 4; ++it) {
            gq[it * 4 * (GPE / 2)] = pack_bf16(p0[0], p1[0]);
            p0 += st0;
            p1 += st1;
        }
    }
    __syncthreads();

    // --- layer 1: H = relu(G @ W1 + b1); wave owns 4 M-tiles x 4 N-tiles
    float b1v[4];
#pragma unroll
    for (int nt = 0; nt < 4; ++nt)
        b1v[nt] = b1[a * HID + (ng * 4 + nt) * 16 + lh];

#pragma unroll
    for (int mm = 0; mm < 4; ++mm) {
        int m = mg * 4 + mm;
        v8f c[4] = {{}, {}, {}, {}};
#pragma unroll
        for (int kf = 0; kf < 4; ++kf) {
            Frag fa;   // A 16x32 bf16: contiguous 16 K per half-wave (half-wave offset +16)
            const unsigned short* gp = G + (m * 16 + lh) * GPE + kf * 32 + hi * 16;
            fa.q[0] = *(const u32x4*)gp;
            fa.q[1] = *(const u32x4*)(gp + 8);
#pragma unroll
            for (int nt = 0; nt < 4; ++nt)
                c[nt] = wmma_bf16(fa.v, Bf[nt][kf].v, c[nt]);
        }
        // epilogue: bias+relu, pack (n, n+16) bf16 pairs into permuted b32 slots
#pragma unroll
        for (int t = 0; t < 2; ++t) {
            int g = ng * 2 + t;                    // 32-column group
#pragma unroll
            for (int r = 0; r < 8; ++r) {          // C: vgpr r -> row m*16 + r + hi*8
                int rr = m * 16 + hi * 8 + r;
                H32[rr * (HPE / 2) + g * 16 + lh] =
                    pack_bf16(fmaxf(c[2 * t][r]     + b1v[2 * t],     0.f),
                              fmaxf(c[2 * t + 1][r] + b1v[2 * t + 1], 0.f));
            }
        }
    }
    __syncthreads();

    // --- layer 2: O = H @ W2 + b2; wave owns rows [wave*16, wave*16+16), cols 0..31
    //     (H storage K-permuted; W2t rows permuted identically in prep, so slots match)
    v8f d0 = {}, d1 = {};
#pragma unroll
    for (int kf = 0; kf < HID / 32; ++kf) {
        Frag fa, fb0, fb1;
        const unsigned short* hp = H + (wave * 16 + lh) * HPE + kf * 32 + hi * 16;
        fa.q[0] = *(const u32x4*)hp;
        fa.q[1] = *(const u32x4*)(hp + 8);
        const unsigned short* p0 = w2 + (0 * 16 + lh) * HID + kf * 32 + hi * 16;
        const unsigned short* p1 = w2 + (1 * 16 + lh) * HID + kf * 32 + hi * 16;
        fb0.q[0] = *(const u32x4*)p0;  fb0.q[1] = *(const u32x4*)(p0 + 8);
        fb1.q[0] = *(const u32x4*)p1;  fb1.q[1] = *(const u32x4*)(p1 + 8);
        d0 = wmma_bf16(fa.v, fb0.v, d0);
        d1 = wmma_bf16(fa.v, fb1.v, d1);
    }

    // --- epilogue: bias + store (out col = a*28 + n, drop padded cols >= 28)
    int col0 = lh;           // always < 28
    int col1 = 16 + lh;      // valid iff lh < 12
    float bb0 = b2[a * DOUT + col0];
    float bb1 = (col1 < DOUT) ? b2[a * DOUT + col1] : 0.f;
#pragma unroll
    for (int r = 0; r < 8; ++r) {
        int row = b0 + wave * 16 + hi * 8 + r;
        out[(size_t)row * NX + a * DOUT + col0] = d0[r] + bb0;
        if (col1 < DOUT)
            out[(size_t)row * NX + a * DOUT + col1] = d1[r] + bb1;
    }
}

// ---------------------------------------------------------------------------
extern "C" void kernel_launch(void* const* d_in, const int* in_sizes, int n_in,
                              void* d_out, int out_size, void* d_ws, size_t ws_size,
                              hipStream_t stream) {
    const float* x      = (const float*)d_in[0];
    const float* u      = (const float*)d_in[1];
    const float* W1     = (const float*)d_in[2];
    const float* b1     = (const float*)d_in[3];
    const float* W2     = (const float*)d_in[4];
    const float* b2     = (const float*)d_in[5];
    const int*   in_idx = (const int*)d_in[6];
    // d_in[7] (out_idx) is the identity scatter over cols 0..895 -> unused
    float* out = (float*)d_out;

    unsigned short* W1t = (unsigned short*)d_ws;                 // 32*256*128 bf16 = 2 MB
    unsigned short* W2t = W1t + (size_t)NA * HID * DIN;          // 32*32*256  bf16 = 512 KB

    prep_w1<<<(NA * HID * DIN) / 256, 256, 0, stream>>>(W1, W1t);
    prep_w2<<<(NA * DOUTP * HID) / 256, 256, 0, stream>>>(W2, W2t);

    size_t smem = 512 + (size_t)BTILE * GPE * 2 + (size_t)BTILE * HPE * 2; // ~100.5 KB
    dim3 grid(B_TOT / BTILE, NA);
    mlp_fused<<<grid, 256, smem, stream>>>(x, u, W1t, b1, W2t, b2, in_idx, out);
}
